// GAL_55542517072401
// MI455X (gfx1250) — matrix operand order repaired and checked by reading, not compiled, not run on hardware
//
#include <hip/hip_runtime.h>
#include <hip/hip_fp16.h>

typedef __attribute__((ext_vector_type(16))) _Float16 v16h;
typedef __attribute__((ext_vector_type(8)))  float    v8f;

#define H_IMG 192
#define W_IMG 192
#define HW    36864
#define C     32
#define WS    16
#define SS    32
#define NH    12
#define NW    12
#define BWIN  144
#define M_TOK 256
#define N_TOK 1024
#define HEADS 4
#define HD    8
#define ATT_SCALE 0.35355339059327373f   // 1/sqrt(8)

#if defined(__HIP_DEVICE_COMPILE__) && __has_builtin(__builtin_amdgcn_global_load_async_to_lds_b128)
#define HAS_ASYNC_LDS 1
typedef int async_i4 __attribute__((ext_vector_type(4)));
typedef __attribute__((address_space(1))) async_i4* g4ptr;   // global int4*
typedef __attribute__((address_space(3))) async_i4* l4ptr;   // LDS int4*
#else
#define HAS_ASYNC_LDS 0
#endif

// ---------------- wave32 reductions ----------------
__device__ __forceinline__ int wred_add(int v) {
  for (int o = 16; o; o >>= 1) v += __shfl_xor(v, o, 32);
  return v;
}
__device__ __forceinline__ float wred_addf(float v) {
  for (int o = 16; o; o >>= 1) v += __shfl_xor(v, o, 32);
  return v;
}
__device__ __forceinline__ float wred_maxf(float v) {
  for (int o = 16; o; o >>= 1) v = fmaxf(v, __shfl_xor(v, o, 32));
  return v;
}
__device__ __forceinline__ int wexcl_scan(int v) {
  int lane = threadIdx.x & 31;
  int s = v;
  for (int o = 1; o < 32; o <<= 1) {
    int t = __shfl_up(s, o, 32);
    if (lane >= o) s += t;
  }
  return s - v;
}

// ---------------- K0: group + l2norm + proj_group + RoPE + per-head l2norm --------
__global__ void k_pre_group(const float* __restrict__ x, const float* __restrict__ Wg,
                            const float* __restrict__ bg, _Float16* __restrict__ XgN,
                            _Float16* __restrict__ qn) {
  __shared__ float sW[C * C];
  __shared__ float sb[C];
  for (int idx = threadIdx.x; idx < C * C; idx += blockDim.x) sW[idx] = Wg[idx];
  if (threadIdx.x < C) sb[threadIdx.x] = bg[threadIdx.x];
  __syncthreads();
  int tid = blockIdx.x * blockDim.x + threadIdx.x;
  int i = tid / W_IMG, j = tid % W_IMG;
  int wh = i / WS, sh = i % WS, ww = j / WS, sw = j % WS;
  int w = wh * NW + ww, m = sh * WS + sw;
  float xv[C];
  float ss = 0.f;
  for (int c = 0; c < C; ++c) { float v = x[c * HW + i * W_IMG + j]; xv[c] = v; ss += v * v; }
  float inv = 1.f / fmaxf(sqrtf(ss), 1e-12f);
  _Float16* xg = XgN + ((size_t)(w * M_TOK + m)) * C;
  for (int c = 0; c < C; ++c) xg[c] = (_Float16)(xv[c] * inv);
  float g[C];
  for (int o = 0; o < C; ++o) {
    float a = sb[o];
    for (int c = 0; c < C; ++c) a += sW[o * C + c] * xv[c];
    g[o] = a;
  }
  _Float16* qo = qn + ((size_t)(w * M_TOK + m)) * C;
  for (int h = 0; h < HEADS; ++h) {
    float r[HD]; float s2 = 0.f;
    for (int p = 0; p < HD / 2; ++p) {
      float fr = powf(10000.f, -(float)p * 0.25f);
      float ang = (float)m * fr;
      float cs = cosf(ang), sn = sinf(ang);
      float e = g[h * HD + 2 * p], od = g[h * HD + 2 * p + 1];
      r[2 * p]     = e * cs - od * sn;
      r[2 * p + 1] = od * cs + e * sn;
    }
    for (int d = 0; d < HD; ++d) s2 += r[d] * r[d];
    float in2 = 1.f / fmaxf(sqrtf(s2), 1e-12f);
    for (int d = 0; d < HD; ++d) qo[h * HD + d] = (_Float16)(r[d] * in2);
  }
}

// ---------------- K1: proj_sample + unfold + l2norm(x) + RoPE(k) + feat ------------
__global__ void k_pre_sample(const float* __restrict__ x, const float* __restrict__ Ws,
                             const float* __restrict__ bs, _Float16* __restrict__ YsN,
                             _Float16* __restrict__ kn, _Float16* __restrict__ fe) {
  __shared__ float sW[2 * C * C];
  __shared__ float sb[2 * C];
  for (int idx = threadIdx.x; idx < 2 * C * C; idx += blockDim.x) sW[idx] = Ws[idx];
  if (threadIdx.x < 2 * C) sb[threadIdx.x] = bs[threadIdx.x];
  __syncthreads();
  int tid = blockIdx.x * blockDim.x + threadIdx.x;
  int w = tid / N_TOK, n = tid % N_TOK;
  int si = n / SS, sj = n % SS;
  int wh = w / NW, ww = w % NW;
  int pi = wh * WS - 8 + si, pj = ww * WS - 8 + sj;
  bool inb = (pi >= 0 && pi < H_IMG && pj >= 0 && pj < W_IMG);
  float xv[C]; float ss = 0.f;
  for (int c = 0; c < C; ++c) {
    float v = inb ? x[c * HW + pi * W_IMG + pj] : 0.f;
    xv[c] = v; ss += v * v;
  }
  float inv = 1.f / fmaxf(sqrtf(ss), 1e-12f);
  _Float16* yo = YsN + ((size_t)tid) * C;
  for (int c = 0; c < C; ++c) yo[c] = (_Float16)(xv[c] * inv);
  float y[2 * C];
  for (int o = 0; o < 2 * C; ++o) {
    float a = sb[o];
    for (int c = 0; c < C; ++c) a += sW[o * C + c] * xv[c];
    y[o] = inb ? a : 0.f;   // unfold zero-pads the *projected* image
  }
  _Float16* fo = fe + ((size_t)tid) * C;
  for (int c = 0; c < C; ++c) fo[c] = (_Float16)y[C + c];
  _Float16* ko = kn + ((size_t)tid) * C;
  for (int h = 0; h < HEADS; ++h) {
    float r[HD]; float s2 = 0.f;
    for (int p = 0; p < HD / 2; ++p) {
      float fr = powf(10000.f, -(float)p * 0.25f);
      float ang = (float)n * fr;
      float cs = cosf(ang), sn = sinf(ang);
      float e = y[h * HD + 2 * p], od = y[h * HD + 2 * p + 1];
      r[2 * p]     = e * cs - od * sn;
      r[2 * p + 1] = od * cs + e * sn;
    }
    for (int d = 0; d < HD; ++d) s2 += r[d] * r[d];
    float in2 = 1.f / fmaxf(sqrtf(s2), 1e-12f);
    for (int d = 0; d < HD; ++d) ko[h * HD + d] = (_Float16)(r[d] * in2);
  }
}

// ---------------- K2a: anti-aliased 2x downsample ([1,3,3,1]/8 separable) ----------
__global__ void k_down(const float* __restrict__ x, float* __restrict__ down) {
  int tid = blockIdx.x * blockDim.x + threadIdx.x;   // 32*96*96
  int c = tid / (96 * 96);
  int r = (tid / 96) % 96;
  int q = tid % 96;
  const float wt[4] = {0.125f, 0.375f, 0.375f, 0.125f};
  float acc = 0.f;
  for (int a = 0; a < 4; ++a) {
    int ii = 2 * r - 1 + a; ii = ii < 0 ? 0 : (ii > H_IMG - 1 ? H_IMG - 1 : ii);
    float rowacc = 0.f;
    for (int b = 0; b < 4; ++b) {
      int jj = 2 * q - 1 + b; jj = jj < 0 ? 0 : (jj > W_IMG - 1 ? W_IMG - 1 : jj);
      rowacc += wt[b] * x[c * HW + ii * W_IMG + jj];
    }
    acc += wt[a] * rowacc;
  }
  down[c * 96 * 96 + r * 96 + q] = acc;
}

// ---------------- K2b: bilinear 2x upsample + abs-diff channel sum -> grouped ------
__global__ void k_diff(const float* __restrict__ x, const float* __restrict__ down,
                       float* __restrict__ Xdg) {
  int tid = blockIdx.x * blockDim.x + threadIdx.x;
  int i = tid / W_IMG, j = tid % W_IMG;
  int k = i >> 1; int i0, i1; float wy0, wy1;
  if ((i & 1) == 0) { i0 = (k - 1 < 0) ? 0 : k - 1; i1 = k; wy0 = 0.25f; wy1 = 0.75f; }
  else              { i0 = k; i1 = (k + 1 > 95) ? 95 : k + 1; wy0 = 0.75f; wy1 = 0.25f; }
  int l = j >> 1; int j0, j1; float wx0, wx1;
  if ((j & 1) == 0) { j0 = (l - 1 < 0) ? 0 : l - 1; j1 = l; wx0 = 0.25f; wx1 = 0.75f; }
  else              { j0 = l; j1 = (l + 1 > 95) ? 95 : l + 1; wx0 = 0.75f; wx1 = 0.25f; }
  float s = 0.f;
  for (int c = 0; c < C; ++c) {
    const float* d = down + c * 96 * 96;
    float up = wy0 * (wx0 * d[i0 * 96 + j0] + wx1 * d[i0 * 96 + j1])
             + wy1 * (wx0 * d[i1 * 96 + j0] + wx1 * d[i1 * 96 + j1]);
    s += fabsf(x[c * HW + i * W_IMG + j] - up);
  }
  int wh = i / WS, sh = i % WS, ww = j / WS, sw = j % WS;
  Xdg[(wh * NW + ww) * M_TOK + sh * WS + sw] = s;
}

// ---------------- K3: 4-iteration proportional budget allocation ------------------
__global__ void k_alloc(const float* __restrict__ Xdg, float* __restrict__ tA) {
  __shared__ float red[256];
  int b = blockIdx.x, t = threadIdx.x;
  float v = Xdg[b * M_TOK + t];
  float rest = 1.f;
  float budget = 255.f * 256.f;   // (TOPK-1)*m
  for (int it = 0; it < 4; ++it) {
    float varp = (rest < 1024.f) ? v : 0.f;
    red[t] = varp; __syncthreads();
    for (int o = 128; o; o >>= 1) { if (t < o) red[t] += red[t + o]; __syncthreads(); }
    float sum = red[0]; __syncthreads();
    float prop = varp / fmaxf(sum, 1e-12f);
    float nr = fminf(fmaxf(rest + rintf(prop * budget), 0.f), 1024.f);
    float delta = nr - rest;
    red[t] = delta; __syncthreads();
    for (int o = 128; o; o >>= 1) { if (t < o) red[t] += red[t + o]; __syncthreads(); }
    float dsum = red[0]; __syncthreads();
    budget = fmaxf(budget - dsum, 0.f);
    rest = nr;
  }
  tA[b * M_TOK + t] = rest;
}

// ---------------- K4: WMMA cosine-sim GEMM + wave32 radix-select -> packed graph ---
__global__ void k_graph(const _Float16* __restrict__ XgN, const _Float16* __restrict__ YsN,
                        const float* __restrict__ tA, unsigned int* __restrict__ graph) {
  extern __shared__ float Drow[];   // 16 * 1024 floats (64 KB dynamic LDS)
  int blk = blockIdx.x;
  int w = blk >> 4, mt = blk & 15;
  int lane = threadIdx.x & 31, wave = threadIdx.x >> 5;
  // A fragment: 16x32 f16 (rows mt*16..mt*16+15, K = channels 0..31)
  v16h a;
  {
    int row = mt * 16 + (lane & 15);
    const _Float16* ap = XgN + ((size_t)(w * M_TOK + row)) * C;
    for (int h = 0; h < 16; ++h) {
      int k = (h < 8 ? h : h + 8) + ((lane & 16) ? 8 : 0);
      a[h] = ap[k];
    }
  }
  for (int t8 = 0; t8 < 8; ++t8) {
    int nt = wave * 8 + t8;
    int col = lane & 15;
    const _Float16* bp = YsN + ((size_t)(w * N_TOK + nt * 16 + col)) * C;
    v16h bf;
    for (int h = 0; h < 16; ++h) {
      int k = (h < 8 ? h : h + 8) + ((lane & 16) ? 8 : 0);
      bf[h] = bp[k];
    }
    v8f cacc = {};
    cacc = __builtin_amdgcn_wmma_f32_16x16x32_f16(false, a, false, bf, (short)0, cacc,
                                                  false, false);
    for (int r = 0; r < 8; ++r) {
      int row = r + ((lane & 16) ? 8 : 0);
      Drow[row * N_TOK + nt * 16 + col] = cacc[r];
    }
  }
  __syncthreads();
  // radix select per row (one wave = one row; 8 waves x 2 rows = 16 rows)
  for (int rr = 0; rr < 2; ++rr) {
    int row = wave * 2 + rr;
    int mrow = mt * 16 + row;
    float t = tA[w * M_TOK + mrow];
    int kkeep = (int)floorf(t) + 1; if (kkeep > N_TOK) kkeep = N_TOK;
    unsigned int key[32];
    for (int i2 = 0; i2 < 32; ++i2) {
      unsigned int u = __float_as_uint(Drow[row * N_TOK + lane * 32 + i2]);
      key[i2] = (u & 0x80000000u) ? ~u : (u | 0x80000000u);   // monotonic key
    }
    unsigned int v = 0;
    for (int bit = 31; bit >= 0; --bit) {
      unsigned int trial = v | (1u << bit);
      int cnt = 0;
      for (int i2 = 0; i2 < 32; ++i2) cnt += (key[i2] >= trial) ? 1 : 0;
      cnt = wred_add(cnt);
      if (cnt >= kkeep) v = trial;   // v -> kkeep-th largest key
    }
    int cg = 0, ce = 0;
    for (int i2 = 0; i2 < 32; ++i2) { cg += (key[i2] > v); ce += (key[i2] == v); }
    int cgTot = wred_add(cg);
    int base = wexcl_scan(ce);
    int needEq = kkeep - cgTot;      // ties kept by smallest index (stable argsort)
    unsigned int word = 0; int eq = 0;
    for (int i2 = 0; i2 < 32; ++i2) {
      bool keep = (key[i2] > v) || ((key[i2] == v) && ((base + eq) < needEq));
      if (key[i2] == v) eq++;
      if (keep) word |= (1u << i2);
    }
    graph[((size_t)(w * M_TOK + mrow)) * 32 + lane] = word;
  }
}

// ---------------- K5: WMMA attention (sim -> masked softmax -> attn@feat) ----------
__global__ void k_attn(const _Float16* __restrict__ qn, const _Float16* __restrict__ kn,
                       const _Float16* __restrict__ fe, const unsigned int* __restrict__ graph,
                       float* __restrict__ aout) {
  extern __shared__ float sim[];    // 16 * 1024 floats (64 KB dynamic LDS)
  __shared__ float rowinv[16];
  __shared__ float outpart[8][16][16];
  __shared__ _Float16 featLds[N_TOK * HD];   // 16 KB: this block's (window, head) V slice
  int blk = blockIdx.x;
  int w = blk >> 6;
  int h = (blk >> 4) & 3;
  int mt = blk & 15;
  int lane = threadIdx.x & 31, wave = threadIdx.x >> 5;

  // ---- stage feat slice to LDS with coalesced 16B rows; overlap with sim GEMM ----
  {
    const _Float16* src = fe + ((size_t)(w * N_TOK)) * C + h * HD;
#if HAS_ASYNC_LDS
    for (int n0 = threadIdx.x; n0 < N_TOK; n0 += 256) {
      __builtin_amdgcn_global_load_async_to_lds_b128(
          (g4ptr)(src + (size_t)n0 * C),
          (l4ptr)&featLds[n0 * HD], 0, 0);
    }
#else
    for (int n0 = threadIdx.x; n0 < N_TOK; n0 += 256) {
      const float4 v4 = *(const float4*)(src + (size_t)n0 * C);
      *(float4*)&featLds[n0 * HD] = v4;
    }
#endif
  }

  // ---- phase 1: sim = l2norm(rope(q)) . l2norm(rope(k))^T, K=8 zero-padded -------
  v16h a;
  {
    int row = mt * 16 + (lane & 15);
    const _Float16* ap = qn + ((size_t)(w * M_TOK + row)) * C + h * HD;
    for (int hh = 0; hh < 16; ++hh) {
      int k = (hh < 8 ? hh : hh + 8) + ((lane & 16) ? 8 : 0);
      a[hh] = (k < HD) ? ap[k] : (_Float16)0.f;
    }
  }
  for (int t8 = 0; t8 < 8; ++t8) {
    int nt = wave * 8 + t8;
    int col = lane & 15;
    const _Float16* bp = kn + ((size_t)(w * N_TOK + nt * 16 + col)) * C + h * HD;
    v16h bf;
    for (int hh = 0; hh < 16; ++hh) {
      int k = (hh < 8 ? hh : hh + 8) + ((lane & 16) ? 8 : 0);
      bf[hh] = (k < HD) ? bp[k] : (_Float16)0.f;
    }
    v8f cacc = {};
    cacc = __builtin_amdgcn_wmma_f32_16x16x32_f16(false, a, false, bf, (short)0, cacc,
                                                  false, false);
    for (int r = 0; r < 8; ++r) {
      int row = r + ((lane & 16) ? 8 : 0);
      sim[row * N_TOK + nt * 16 + col] = cacc[r];
    }
  }
  __syncthreads();

  // ---- phase 2: masked softmax, one wave per row; graph word index == lane -------
  for (int rr = 0; rr < 2; ++rr) {
    int row = wave * 2 + rr;
    int mrow = mt * 16 + row;
    unsigned int g = graph[((size_t)(w * M_TOK + mrow)) * 32 + lane];
    float lv[32];
    float lmax = -1e30f;
    for (int i2 = 0; i2 < 32; ++i2) {
      float bitv = (float)((g >> i2) & 1u);
      float lg = (sim[row * N_TOK + lane * 32 + i2] - 100.f * (1.f - bitv)) * ATT_SCALE;
      lv[i2] = lg;
      lmax = fmaxf(lmax, lg);
    }
    lmax = wred_maxf(lmax);
    float ssum = 0.f;
    for (int i2 = 0; i2 < 32; ++i2) {
      float p = expf(lv[i2] - lmax);
      sim[row * N_TOK + lane * 32 + i2] = p;
      ssum += p;
    }
    ssum = wred_addf(ssum);
    if (lane == 0) rowinv[row] = 1.f / ssum;
  }
#if HAS_ASYNC_LDS
  asm volatile("s_wait_asynccnt 0x0" ::: "memory");   // retire this wave's async copies
#endif
  __syncthreads();

  // ---- phase 3: out = attn @ feat : waves split K=1024 into 8 x (4 x 32) ---------
  v8f acc = {};
  for (int t4 = 0; t4 < 4; ++t4) {
    int kt = wave * 4 + t4;
    v16h af;
    {
      int row = lane & 15;
      for (int hh = 0; hh < 16; ++hh) {
        int k = (hh < 8 ? hh : hh + 8) + ((lane & 16) ? 8 : 0);
        af[hh] = (_Float16)sim[row * N_TOK + kt * 32 + k];
      }
    }
    v16h bf;
    {
      int col = lane & 15;
      for (int hh = 0; hh < 16; ++hh) {
        int k = (hh < 8 ? hh : hh + 8) + ((lane & 16) ? 8 : 0);
        bf[hh] = (col < HD) ? featLds[(kt * 32 + k) * HD + col] : (_Float16)0.f;
      }
    }
    acc = __builtin_amdgcn_wmma_f32_16x16x32_f16(false, af, false, bf, (short)0, acc,
                                                 false, false);
  }
  {
    int col = lane & 15;
    for (int r = 0; r < 8; ++r) {
      int row = r + ((lane & 16) ? 8 : 0);
      outpart[wave][row][col] = acc[r];
    }
  }
  __syncthreads();
  {
    int row = threadIdx.x >> 4;   // 0..15
    int col = threadIdx.x & 15;
    float s = 0.f;
    for (int wv = 0; wv < 8; ++wv) s += outpart[wv][row][col];
    if (col < HD)
      aout[((size_t)(w * M_TOK + mt * 16 + row)) * C + h * HD + col] = s * rowinv[row];
  }
}

// ---------------- K6: output projection Wp, regroup to NCHW image ------------------
__global__ void k_proj(const float* __restrict__ aout, const float* __restrict__ Wp,
                       const float* __restrict__ bp, float* __restrict__ img) {
  __shared__ float sW[C * C];
  __shared__ float sb[C];
  for (int idx = threadIdx.x; idx < C * C; idx += blockDim.x) sW[idx] = Wp[idx];
  if (threadIdx.x < C) sb[threadIdx.x] = bp[threadIdx.x];
  __syncthreads();
  int tid = blockIdx.x * blockDim.x + threadIdx.x;
  int i = tid / W_IMG, j = tid % W_IMG;
  int wh = i / WS, sh = i % WS, ww = j / WS, sw = j % WS;
  const float* ov = aout + ((size_t)((wh * NW + ww) * M_TOK + sh * WS + sw)) * C;
  float xv[C];
  for (int c = 0; c < C; ++c) xv[c] = ov[c];
  for (int o = 0; o < C; ++o) {
    float acc2 = sb[o];
    for (int c = 0; c < C; ++c) acc2 += sW[o * C + c] * xv[c];
    img[o * HW + i * W_IMG + j] = acc2;
  }
}

// ---------------- K7: 3x3 conv (SAME) + channel LayerNorm + residual ---------------
__global__ void k_conv_ln(const float* __restrict__ img, const float* __restrict__ Wc,
                          const float* __restrict__ bc, const float* __restrict__ gamma,
                          const float* __restrict__ beta, float* __restrict__ outp) {
  __shared__ float sW[C * C * 9];
  __shared__ float sbc[C], sg[C], sbt[C];
  for (int idx = threadIdx.x; idx < C * C * 9; idx += blockDim.x) sW[idx] = Wc[idx];
  if (threadIdx.x < C) {
    sbc[threadIdx.x] = bc[threadIdx.x];
    sg[threadIdx.x]  = gamma[threadIdx.x];
    sbt[threadIdx.x] = beta[threadIdx.x];
  }
  __syncthreads();
  int tid = blockIdx.x * blockDim.x + threadIdx.x;
  int i = tid / W_IMG, j = tid % W_IMG;
  __builtin_prefetch(&img[i * W_IMG + j], 0, 1);   // global_prefetch_b8
  float acc[C];
  for (int o = 0; o < C; ++o) acc[o] = sbc[o];
  for (int c = 0; c < C; ++c) {
    float nb[9];
    for (int dy = 0; dy < 3; ++dy)
      for (int dx = 0; dx < 3; ++dx) {
        int ii = i + dy - 1, jj = j + dx - 1;
        nb[dy * 3 + dx] = (ii >= 0 && ii < H_IMG && jj >= 0 && jj < W_IMG)
                            ? img[c * HW + ii * W_IMG + jj] : 0.f;
      }
    for (int o = 0; o < C; ++o) {
      const float* wp = sW + (o * C + c) * 9;
      float a = acc[o];
      for (int q = 0; q < 9; ++q) a += wp[q] * nb[q];
      acc[o] = a;
    }
  }
  float mu = 0.f;
  for (int o = 0; o < C; ++o) mu += acc[o];
  mu *= (1.f / C);
  float var = 0.f;
  for (int o = 0; o < C; ++o) { float d = acc[o] - mu; var += d * d; }
  var *= (1.f / C);
  float rs = rsqrtf(var + 1e-5f);
  for (int o = 0; o < C; ++o) {
    float nv = (acc[o] - mu) * rs * sg[o] + sbt[o];
    outp[o * HW + i * W_IMG + j] = img[o * HW + i * W_IMG + j] + nv;
  }
}

// ---------------- launcher ---------------------------------------------------------
extern "C" void kernel_launch(void* const* d_in, const int* in_sizes, int n_in,
                              void* d_out, int out_size, void* d_ws, size_t ws_size,
                              hipStream_t stream) {
  (void)in_sizes; (void)n_in; (void)out_size; (void)ws_size;
  const float* x     = (const float*)d_in[0];
  const float* Wg    = (const float*)d_in[1];
  const float* bg    = (const float*)d_in[2];
  const float* Ws    = (const float*)d_in[3];
  const float* bs    = (const float*)d_in[4];
  const float* Wp    = (const float*)d_in[5];
  const float* bp    = (const float*)d_in[6];
  const float* Wc    = (const float*)d_in[7];
  const float* bc    = (const float*)d_in[8];
  const float* gamma = (const float*)d_in[9];
  const float* beta  = (const float*)d_in[10];
  float* outF = (float*)d_out;

  char* ws = (char*)d_ws;
  size_t off = 0;
  auto alloc = [&](size_t bytes) -> void* {
    void* p = ws + off;
    off = (off + bytes + 255) & ~(size_t)255;
    return p;
  };
  float*        Xdg  = (float*)alloc((size_t)BWIN * M_TOK * 4);
  float*        tA   = (float*)alloc((size_t)BWIN * M_TOK * 4);
  float*        down = (float*)alloc((size_t)C * 96 * 96 * 4);
  _Float16*     XgN  = (_Float16*)alloc((size_t)BWIN * M_TOK * C * 2);
  _Float16*     YsN  = (_Float16*)alloc((size_t)BWIN * N_TOK * C * 2);
  _Float16*     qn   = (_Float16*)alloc((size_t)BWIN * M_TOK * C * 2);
  _Float16*     kn   = (_Float16*)alloc((size_t)BWIN * N_TOK * C * 2);
  _Float16*     fe   = (_Float16*)alloc((size_t)BWIN * N_TOK * C * 2);
  unsigned int* grph = (unsigned int*)alloc((size_t)BWIN * M_TOK * 32 * 4);
  float*        aout = (float*)alloc((size_t)BWIN * M_TOK * C * 4);
  float*        img  = (float*)alloc((size_t)C * HW * 4);

  k_pre_group <<<HW / 256, 256, 0, stream>>>(x, Wg, bg, XgN, qn);
  k_pre_sample<<<BWIN * N_TOK / 256, 256, 0, stream>>>(x, Ws, bs, YsN, kn, fe);
  k_down      <<<C * 96 * 96 / 256, 256, 0, stream>>>(x, down);
  k_diff      <<<HW / 256, 256, 0, stream>>>(x, down, Xdg);
  k_alloc     <<<BWIN, 256, 0, stream>>>(Xdg, tA);
  k_graph     <<<BWIN * 16, 256, 16 * N_TOK * 4, stream>>>(XgN, YsN, tA, grph);
  k_attn      <<<BWIN * HEADS * 16, 256, 16 * N_TOK * 4, stream>>>(qn, kn, fe, grph, aout);
  k_proj      <<<HW / 256, 256, 0, stream>>>(aout, Wp, bp, img);
  k_conv_ln   <<<HW / 256, 256, 0, stream>>>(img, Wc, bc, gamma, beta, outF);
}
